// GModule_80685255622988
// MI455X (gfx1250) — matrix operand order repaired and checked by reading, not compiled, not run on hardware
//
#include <hip/hip_runtime.h>

// ---------------------------------------------------------------------------
// MI455X (gfx1250) implementation of the graph-matching forward pass.
// All GEMMs run on v_wmma_f32_16x16x32_bf16 (wave32 WMMA, fp32 accumulate).
// LayerNorm / softmax / instance-norm statistics stay in fp32 VALU.
// ---------------------------------------------------------------------------

typedef __attribute__((ext_vector_type(16))) __bf16 v16bf;
typedef __attribute__((ext_vector_type(8)))  float  v8f;

#define DD 256
#define NN 4096

// round-to-nearest-even f32 -> bf16 (bit pattern in ushort)
__device__ __forceinline__ unsigned short f2bf(float f) {
  unsigned u = __float_as_uint(f);
  u += 0x7FFFu + ((u >> 16) & 1u);
  return (unsigned short)(u >> 16);
}

union FragU { v16bf v; uint4 q[2]; };

// A operand (16x32, bf16). Memory: A[M,K] row-major (K contiguous).
// ISA layout: lanes 0-15 hold M=lane, halves0-7 = K0..7, halves8-15 = K16..23;
//             lanes 16-31 same rows, K8..15 / K24..31.
__device__ __forceinline__ v16bf load_a_frag(const unsigned short* __restrict__ A,
                                             int K, int row, int k0, int lh) {
  const unsigned short* p = A + (size_t)row * K + k0 + lh * 8;
  FragU f;
  f.q[0] = *reinterpret_cast<const uint4*>(p);
  f.q[1] = *reinterpret_cast<const uint4*>(p + 16);
  return f.v;
}

// B operand (32x16, bf16). Memory: B[N,K] row-major (so B_op[k][n] = B[n][k]).
// ISA layout: lanes 0-15 (N=lane) hold K0..7 / K16..23; lanes 16-31 K8..15 / K24..31.
__device__ __forceinline__ v16bf load_b_frag(const unsigned short* __restrict__ B,
                                             int K, int col, int k0, int lh) {
  const unsigned short* p = B + (size_t)col * K + k0 + lh * 8;
  FragU f;
  f.q[0] = *reinterpret_cast<const uint4*>(p);
  f.q[1] = *reinterpret_cast<const uint4*>(p + 16);
  return f.v;
}

// ---------------------------------------------------------------------------
// Universal GEMM: C[M,N] = (A[M,K] * B[N,K]^T + bias[N]) * alpha
// A,B bf16 row-major (K contiguous). Output fp32 or bf16 (compile-time).
// Block = 256 threads = 8 waves arranged 4(M) x 2(N).
// Wave tile = (MI*16) x (NJ*16); block tile = (4*MI*16) x (2*NJ*16).
//   - Big GEMMs (N=4096): MI=NJ=4 -> 64x64 wave tile, 16 loads : 16 WMMAs.
//   - Small GEMMs (N=256): MI=NJ=2 -> 32x32 wave tile, more blocks in flight.
// C VGPR layout: acc[r], lanes0-15 -> (M=r, N=lane); lanes16-31 -> (M=r+8).
// ---------------------------------------------------------------------------
template <int MI, int NJ, bool HAS_BIAS, bool OUT_BF16>
__global__ __launch_bounds__(256)
void gemm_bf16_kernel(const unsigned short* __restrict__ A,
                      const unsigned short* __restrict__ B,
                      void* __restrict__ Cout,
                      const float* __restrict__ bias,
                      float alpha, int M, int N, int K)
{
  const int lane = threadIdx.x & 31;
  const int wave = threadIdx.x >> 5;
  const int wm = wave >> 1;                 // 0..3
  const int wn = wave & 1;                  // 0..1
  const int bm = blockIdx.y * (4 * MI * 16) + wm * (MI * 16);
  const int bn = blockIdx.x * (2 * NJ * 16) + wn * (NJ * 16);
  const int lh = lane >> 4;
  const int ll = lane & 15;

  v8f acc[MI][NJ];
#pragma unroll
  for (int i = 0; i < MI; i++)
#pragma unroll
    for (int j = 0; j < NJ; j++)
#pragma unroll
      for (int e = 0; e < 8; e++) acc[i][j][e] = 0.0f;

  for (int k0 = 0; k0 < K; k0 += 32) {
    v16bf af[MI], bfr[NJ];
#pragma unroll
    for (int i = 0; i < MI; i++) af[i] = load_a_frag(A, K, bm + i * 16 + ll, k0, lh);
#pragma unroll
    for (int j = 0; j < NJ; j++) bfr[j] = load_b_frag(B, K, bn + j * 16 + ll, k0, lh);

    // Unconditional speculative prefetch of the next K-tile (global_prefetch_b8);
    // branch-free: a past-the-end prefetch on the last iteration is silently
    // dropped by the speculative-prefetch path.
    __builtin_prefetch(A + (size_t)(bm + ll + lh * 16) * K + k0 + 32, 0, 1);
    __builtin_prefetch(B + (size_t)(bn + ll + lh * 16) * K + k0 + 32, 0, 1);

#pragma unroll
    for (int i = 0; i < MI; i++)
#pragma unroll
      for (int j = 0; j < NJ; j++)
        acc[i][j] = __builtin_amdgcn_wmma_f32_16x16x32_bf16(
            false, af[i], false, bfr[j], (short)0, acc[i][j], false, false);
  }

#pragma unroll
  for (int j = 0; j < NJ; j++) {
    const int col = bn + j * 16 + ll;
    float bv = 0.0f;
    if (HAS_BIAS) bv = bias[col];
#pragma unroll
    for (int i = 0; i < MI; i++) {
#pragma unroll
      for (int r = 0; r < 8; r++) {
        const int row = bm + i * 16 + r + lh * 8;
        const float v = (acc[i][j][r] + bv) * alpha;
        if (OUT_BF16) ((unsigned short*)Cout)[(size_t)row * N + col] = f2bf(v);
        else          ((float*)Cout)[(size_t)row * N + col] = v;
      }
    }
  }
}

// ---------------------------------------------------------------------------
// Row LayerNorm over 256 cols with bias add and optional ReLU; bf16 out.
// One block (256 threads) per row.
// ---------------------------------------------------------------------------
__global__ __launch_bounds__(256)
void ln_bias_kernel(const float* __restrict__ X, const float* __restrict__ bias,
                    unsigned short* __restrict__ out, int relu)
{
  __shared__ float sh[256];
  const int row = blockIdx.x;
  const int t = threadIdx.x;
  float x = X[(size_t)row * DD + t] + bias[t];
  sh[t] = x; __syncthreads();
  for (int s = 128; s > 0; s >>= 1) { if (t < s) sh[t] += sh[t + s]; __syncthreads(); }
  const float mean = sh[0] / (float)DD; __syncthreads();
  const float d = x - mean;
  sh[t] = d * d; __syncthreads();
  for (int s = 128; s > 0; s >>= 1) { if (t < s) sh[t] += sh[t + s]; __syncthreads(); }
  const float var = sh[0] / (float)DD;
  float y = d * rsqrtf(var + 1e-5f);
  if (relu) y = fmaxf(y, 0.0f);
  out[(size_t)row * DD + t] = f2bf(y);
}

// ---------------------------------------------------------------------------
// Row softmax over 4096 cols, fp32 in -> bf16 out IN PLACE (bf16 packed densely,
// row stride 4096, from the start of S). One block per row; a block's reads complete before
// its own writes (block barriers only; no ordering across blocks).
// ---------------------------------------------------------------------------
__global__ __launch_bounds__(256)
void softmax_rows_kernel(float* __restrict__ S)
{
  __shared__ float sh[256];
  const int row = blockIdx.x;
  const int t = threadIdx.x;
  const float* xr = S + (size_t)row * NN;
  float xv[16];
  float m = -3.4e38f;
#pragma unroll
  for (int j = 0; j < 16; j++) { xv[j] = xr[t + j * 256]; m = fmaxf(m, xv[j]); }
  sh[t] = m; __syncthreads();
  for (int s = 128; s > 0; s >>= 1) { if (t < s) sh[t] = fmaxf(sh[t], sh[t + s]); __syncthreads(); }
  m = sh[0]; __syncthreads();
  float sum = 0.0f;
#pragma unroll
  for (int j = 0; j < 16; j++) { xv[j] = expf(xv[j] - m); sum += xv[j]; }
  sh[t] = sum; __syncthreads();
  for (int s = 128; s > 0; s >>= 1) { if (t < s) sh[t] += sh[t + s]; __syncthreads(); }
  const float inv = 1.0f / sh[0];
  __syncthreads();   // ensure every lane's fp32 reads are done before bf16 writes
  unsigned short* pr = (unsigned short*)S + (size_t)row * NN;
#pragma unroll
  for (int j = 0; j < 16; j++) pr[t + j * 256] = f2bf(xv[j] * inv);
}

// bf16 [R,C] -> [C,R] transpose via LDS tile
__global__ __launch_bounds__(256)
void transpose_bf16_kernel(const unsigned short* __restrict__ in,
                           unsigned short* __restrict__ out, int R, int C)
{
  __shared__ unsigned short tile[32][33];
  const int bx = blockIdx.x * 32;   // input col base
  const int by = blockIdx.y * 32;   // input row base
  const int tx = threadIdx.x, ty = threadIdx.y;
#pragma unroll
  for (int j = 0; j < 32; j += 8)
    tile[ty + j][tx] = in[(size_t)(by + ty + j) * C + bx + tx];
  __syncthreads();
#pragma unroll
  for (int j = 0; j < 32; j += 8)
    out[(size_t)(bx + ty + j) * R + by + tx] = tile[tx][ty + j];
}

__global__ void cast_bf16_kernel(const float* __restrict__ x,
                                 unsigned short* __restrict__ y, int n)
{
  const int i = blockIdx.x * blockDim.x + threadIdx.x;
  if (i < n) y[i] = f2bf(x[i]);
}

// Ab = bf16( 0.5*(A + A^T) ), 256x256
__global__ __launch_bounds__(256)
void asym_bf16_kernel(const float* __restrict__ A, unsigned short* __restrict__ y)
{
  const int i = blockIdx.x, j = threadIdx.x;
  y[i * DD + j] = f2bf(0.5f * (A[i * DD + j] + A[j * DD + i]));
}

// deterministic two-stage whole-matrix mean/var reduction
__global__ __launch_bounds__(256)
void reduce_partial_kernel(const float* __restrict__ x, float* __restrict__ partial,
                           long long n)
{
  __shared__ float s1[256], s2[256];
  const int t = threadIdx.x;
  float a = 0.0f, b = 0.0f;
  for (long long i = (long long)blockIdx.x * 256 + t; i < n; i += (long long)gridDim.x * 256) {
    const float v = x[i]; a += v; b += v * v;
  }
  s1[t] = a; s2[t] = b; __syncthreads();
  for (int s = 128; s > 0; s >>= 1) {
    if (t < s) { s1[t] += s1[t + s]; s2[t] += s2[t + s]; }
    __syncthreads();
  }
  if (t == 0) { partial[2 * blockIdx.x] = s1[0]; partial[2 * blockIdx.x + 1] = s2[0]; }
}

__global__ __launch_bounds__(256)
void reduce_final_kernel(const float* __restrict__ partial, float* __restrict__ red,
                         int nparts, float n)
{
  __shared__ float s1[256], s2[256];
  const int t = threadIdx.x;
  float a = 0.0f, b = 0.0f;
  for (int i = t; i < nparts; i += 256) { a += partial[2 * i]; b += partial[2 * i + 1]; }
  s1[t] = a; s2[t] = b; __syncthreads();
  for (int s = 128; s > 0; s >>= 1) {
    if (t < s) { s1[t] += s1[t + s]; s2[t] += s2[t + s]; }
    __syncthreads();
  }
  if (t == 0) {
    const float mean = s1[0] / n;
    const float var  = s2[0] / n - mean * mean;
    red[0] = mean;
    red[1] = rsqrtf(var + 1e-5f);
  }
}

__global__ __launch_bounds__(256)
void instnorm_kernel(float* __restrict__ x, const float* __restrict__ red, long long n)
{
  const float mean = red[0], inv = red[1];
  for (long long i = (long long)blockIdx.x * blockDim.x + threadIdx.x; i < n;
       i += (long long)gridDim.x * blockDim.x)
    x[i] = (x[i] - mean) * inv;
}

// ---------------------------------------------------------------------------
// Host orchestration
// ---------------------------------------------------------------------------
extern "C" void kernel_launch(void* const* d_in, const int* in_sizes, int n_in,
                              void* d_out, int out_size, void* d_ws, size_t ws_size,
                              hipStream_t stream)
{
  (void)in_sizes; (void)n_in; (void)out_size; (void)ws_size;
  const int D = DD, N = NN;

  const float* nodes1 = (const float*)d_in[0];
  const float* nodes2 = (const float*)d_in[1];
  const float* Wf32[10] = {
    (const float*)d_in[2],  (const float*)d_in[4],                 // W_in1, W_in2
    (const float*)d_in[6],  (const float*)d_in[8],                 // Wq_i, Wk_i
    (const float*)d_in[10], (const float*)d_in[12],                // Wv_i, Wo_i
    (const float*)d_in[14], (const float*)d_in[16],                // Wq_c, Wk_c
    (const float*)d_in[18], (const float*)d_in[20]                 // Wv_c, Wo_c
  };
  const float* b_in1 = (const float*)d_in[3];
  const float* b_in2 = (const float*)d_in[5];
  const float* bq_i = (const float*)d_in[7],  *bk_i = (const float*)d_in[9];
  const float* bv_i = (const float*)d_in[11], *bo_i = (const float*)d_in[13];
  const float* bq_c = (const float*)d_in[15], *bk_c = (const float*)d_in[17];
  const float* bv_c = (const float*)d_in[19], *bo_c = (const float*)d_in[21];
  const float* Aaff = (const float*)d_in[22];

  // ---- workspace carve-out (~24 MB) ----
  char* wsp = (char*)d_ws;
  size_t off = 0;
  auto alloc = [&](size_t bytes) -> void* {
    void* p = wsp + off;
    off = (off + bytes + 255) & ~(size_t)255;
    return p;
  };
  unsigned short* Wb[10];
  for (int i = 0; i < 10; i++) Wb[i] = (unsigned short*)alloc((size_t)D * D * 2);
  unsigned short* Ab  = (unsigned short*)alloc((size_t)D * D * 2);
  unsigned short* Xb  = (unsigned short*)alloc((size_t)N * D * 2);  // generic bf16 scratch
  unsigned short* E1  = (unsigned short*)alloc((size_t)N * D * 2);
  unsigned short* E2  = (unsigned short*)alloc((size_t)N * D * 2);
  unsigned short* C1  = (unsigned short*)alloc((size_t)N * D * 2);
  unsigned short* C2  = (unsigned short*)alloc((size_t)N * D * 2);
  unsigned short* Qb  = (unsigned short*)alloc((size_t)N * D * 2);
  unsigned short* Kb  = (unsigned short*)alloc((size_t)N * D * 2);
  unsigned short* Vb  = (unsigned short*)alloc((size_t)N * D * 2);
  unsigned short* VTb = (unsigned short*)alloc((size_t)N * D * 2);  // [D, N]
  float* Tf      = (float*)alloc((size_t)N * D * 4);
  float* partial = (float*)alloc(1024 * 2 * 4);
  float* red     = (float*)alloc(2 * 4);

  // compile-time-dispatched GEMM launcher
  auto gemm = [&](const unsigned short* A, const unsigned short* B, void* C,
                  const float* bias, float alpha, int M, int Nn, int K, int outBf16) {
    dim3 block(256);
    if (Nn >= 1024) {
      dim3 grid(Nn / 128, M / 256);   // MI=NJ=4: block tile 256x128
      if (bias) {
        if (outBf16) gemm_bf16_kernel<4,4,true, true ><<<grid, block, 0, stream>>>(A,B,C,bias,alpha,M,Nn,K);
        else         gemm_bf16_kernel<4,4,true, false><<<grid, block, 0, stream>>>(A,B,C,bias,alpha,M,Nn,K);
      } else {
        if (outBf16) gemm_bf16_kernel<4,4,false,true ><<<grid, block, 0, stream>>>(A,B,C,bias,alpha,M,Nn,K);
        else         gemm_bf16_kernel<4,4,false,false><<<grid, block, 0, stream>>>(A,B,C,bias,alpha,M,Nn,K);
      }
    } else {
      dim3 grid(Nn / 64, M / 128);    // MI=NJ=2: block tile 128x64
      if (bias) {
        if (outBf16) gemm_bf16_kernel<2,2,true, true ><<<grid, block, 0, stream>>>(A,B,C,bias,alpha,M,Nn,K);
        else         gemm_bf16_kernel<2,2,true, false><<<grid, block, 0, stream>>>(A,B,C,bias,alpha,M,Nn,K);
      } else {
        if (outBf16) gemm_bf16_kernel<2,2,false,true ><<<grid, block, 0, stream>>>(A,B,C,bias,alpha,M,Nn,K);
        else         gemm_bf16_kernel<2,2,false,false><<<grid, block, 0, stream>>>(A,B,C,bias,alpha,M,Nn,K);
      }
    }
  };
  auto cast = [&](const float* x, unsigned short* y, int n) {
    cast_bf16_kernel<<<dim3((n + 255) / 256), dim3(256), 0, stream>>>(x, y, n);
  };

  // ---- precompute bf16 weights + symmetric affinity ----
  for (int i = 0; i < 10; i++) cast(Wf32[i], Wb[i], D * D);
  asym_bf16_kernel<<<dim3(D), dim3(D), 0, stream>>>(Aaff, Ab);

  // ---- head_in: LN(x W1^T + b1) -> ReLU -> LN(h W2^T + b2) ----
  auto head_in = [&](const float* X, unsigned short* E) {
    cast(X, Xb, N * D);
    gemm(Xb, Wb[0], Tf, nullptr, 1.0f, N, D, D, 0);
    ln_bias_kernel<<<dim3(N), dim3(256), 0, stream>>>(Tf, b_in1, Xb, 1);
    gemm(Xb, Wb[1], Tf, nullptr, 1.0f, N, D, D, 0);
    ln_bias_kernel<<<dim3(N), dim3(256), 0, stream>>>(Tf, b_in2, E, 0);
  };
  head_in(nodes1, E1);
  head_in(nodes2, E2);

  // ---- single-head attention; scores (64 MB fp32) live in d_out ----
  auto mha = [&](const unsigned short* Eq, const unsigned short* Ekv,
                 const unsigned short* Wq, const unsigned short* Wk,
                 const unsigned short* Wv, const unsigned short* Wo,
                 const float* bq, const float* bk, const float* bv, const float* bo,
                 unsigned short* Eout) {
    gemm(Eq,  Wq, Qb, bq, 0.0625f, N, D, D, 1);         // Q pre-scaled by 1/sqrt(256)
    gemm(Ekv, Wk, Kb, bk, 1.0f,    N, D, D, 1);
    gemm(Ekv, Wv, Vb, bv, 1.0f,    N, D, D, 1);
    transpose_bf16_kernel<<<dim3(D / 32, N / 32), dim3(32, 8), 0, stream>>>(Vb, VTb, N, D);
    gemm(Qb, Kb, d_out, nullptr, 1.0f, N, N, D, 0);     // S = Q K^T  (fp32, in d_out)
    softmax_rows_kernel<<<dim3(N), dim3(256), 0, stream>>>((float*)d_out);  // -> bf16 P in place
    gemm((const unsigned short*)d_out, VTb, Xb, nullptr, 1.0f, N, D, N, 1); // O = P V
    gemm(Xb, Wo, Eout, bo, 1.0f, N, D, D, 1);           // out = O Wo^T + bo
  };

  mha(E1, E1, Wb[2], Wb[3], Wb[4], Wb[5], bq_i, bk_i, bv_i, bo_i, E1);
  mha(E2, E2, Wb[2], Wb[3], Wb[4], Wb[5], bq_i, bk_i, bv_i, bo_i, E2);
  mha(E1, E2, Wb[6], Wb[7], Wb[8], Wb[9], bq_c, bk_c, bv_c, bo_c, C1);
  mha(E2, E1, Wb[6], Wb[7], Wb[8], Wb[9], bq_c, bk_c, bv_c, bo_c, C2);

  // ---- M = (n1c @ Asym) @ n2c^T ; Asym symmetric so Asym^T == Asym ----
  gemm(C1, Ab, Xb, nullptr, 1.0f, N, D, D, 1);
  gemm(Xb, C2, d_out, nullptr, 1.0f, N, N, D, 0);

  // ---- whole-matrix instance norm ----
  const long long total = (long long)N * N;
  reduce_partial_kernel<<<dim3(1024), dim3(256), 0, stream>>>((const float*)d_out, partial, total);
  reduce_final_kernel<<<dim3(1), dim3(256), 0, stream>>>(partial, red, 1024, (float)total);
  instnorm_kernel<<<dim3(4096), dim3(256), 0, stream>>>((float*)d_out, red, total);
}
